// MambaBlock_38766374814391
// MI455X (gfx1250) — matrix operand order repaired and checked by reading, not compiled, not run on hardware
//
#include <hip/hip_runtime.h>
#include <hip/hip_bf16.h>

// ---------------------------------------------------------------------------
// Mamba block for MI455X (gfx1250).
// bf16 WMMA GEMMs, fully-async double-buffered LDS staging, fused epilogues;
// VALU kernels for LN / conv / selective scan.
// ---------------------------------------------------------------------------

typedef __attribute__((ext_vector_type(16))) __bf16 v16bf;
typedef __attribute__((ext_vector_type(8)))  __bf16 v8bf;
typedef __attribute__((ext_vector_type(2)))  __bf16 v2bf;
typedef __attribute__((ext_vector_type(8)))  float  v8f;
typedef __attribute__((ext_vector_type(4)))  int    v4i;

static constexpr int kB   = 2;
static constexpr int kL   = 2048;
static constexpr int kDM  = 1024;     // D_MODEL
static constexpr int kDI  = 2048;     // D_INNER
static constexpr int kDS  = 16;       // D_STATE
static constexpr int kDTR = 64;       // DT_RANK
static constexpr int kDF  = 4096;     // D_FF
static constexpr int kBL  = kB * kL;  // 4096 rows
static constexpr int kNP  = 128;      // padded width for the 96-wide x-proj

// ---------------- feature probes ----------------
#if defined(__has_builtin)
#if __has_builtin(__builtin_amdgcn_cvt_pk_bf16_f32)
#define HAVE_CVT_PK 1
#endif
#if __has_builtin(__builtin_amdgcn_global_load_async_to_lds_b128)
#define HAVE_ASYNC_LDS 1
#endif
#if __has_builtin(__builtin_amdgcn_s_wait_asynccnt)
#define HAVE_WAIT_ASYNC 1
#endif
#endif

template <int Ncnt>
__device__ __forceinline__ void wait_async() {
#if defined(HAVE_ASYNC_LDS)
#if defined(HAVE_WAIT_ASYNC)
    __builtin_amdgcn_s_wait_asynccnt(Ncnt);
#else
    asm volatile("s_wait_asynccnt %0" :: "i"(Ncnt) : "memory");
#endif
#endif
}

// ---------------- bf16 helpers (native cvt) ----------------
__device__ __forceinline__ unsigned pk2bf(float a, float b) {
#if defined(HAVE_CVT_PK)
    v2bf r = __builtin_amdgcn_cvt_pk_bf16_f32(a, b);
    union { v2bf v; unsigned u; } q; q.v = r; return q.u;
#else
    union { __bf16 h[2]; unsigned u; } q;
    q.h[0] = (__bf16)a; q.h[1] = (__bf16)b;
    return q.u;
#endif
}
__device__ __forceinline__ unsigned short bf16bits(float f) {
    union { __bf16 h; unsigned short s; } q; q.h = (__bf16)f; return q.s;
}

// ---------------- f32 -> bf16 bulk converter (n % 4 == 0) ----------------
__global__ __launch_bounds__(256)
void cvt_bf16_k(const float* __restrict__ in, unsigned short* __restrict__ out, int n)
{
    const int i = (blockIdx.x * 256 + threadIdx.x) * 4;
    if (i >= n) return;
    const float4 v = *(const float4*)&in[i];
    uint2 p;
    p.x = pk2bf(v.x, v.y);
    p.y = pk2bf(v.z, v.w);
    *(uint2*)&out[i] = p;
}

// ---------------- WMMA GEMM: D[M,N] = A[M,K] * W[N,K]^T (+ epilogue) -------
// Requirements: M % 128 == 0, N % 128 == 0, K % 32 == 0 (callers guarantee).
enum { EPI_NONE = 0, EPI_BIAS_SOFTPLUS = 1, EPI_RELU = 2, EPI_ADD_RES = 3 };

#define BM 128
#define BN 128
#define BK 32
#define ST 48   // bf16 elements per LDS row (32 + 16 pad) -> 96B, 16B aligned

template <int EPI, bool ABF16, bool OUTBF16>
__global__ __launch_bounds__(256, 2)
void gemm_wmma(const void* __restrict__ Araw, int lda,
               const unsigned short* __restrict__ Wb,   // bf16 weights [N,K]
               const float* __restrict__ aux,           // bias[N] or res[M*N]
               void* __restrict__ outRaw,
               int M, int N, int K)
{
    __shared__ unsigned short As[2][BM * ST];
    __shared__ unsigned short Ws[2][BN * ST];

    const int tid  = threadIdx.x;
    const int lane = tid & 31;
    const int wave = tid >> 5;      // 0..7
    const int wm   = wave >> 1;     // 0..3  (32 rows each)
    const int wn   = wave & 1;      // 0..1  (64 cols each)
    const int lrow = lane & 15;
    const int half = lane >> 4;

    const int bM = blockIdx.y * BM;
    const int bN = blockIdx.x * BN;

    const int chunk = tid & 3;      // which 16B chunk in a 64B bf16 row
    const int crow  = tid >> 2;     // 0..63
    const int slot  = tid & 7;      // f32 path: group of 4 floats
    const int srow  = tid >> 3;     // 0..31

    // per-thread async copies issued per stage (A: 2 if bf16; W: always 2)
    constexpr int NASYNC = ABF16 ? 4 : 2;

    v8f acc[2][4] = {};

    auto stage = [&](int k0, int buf) {
        if constexpr (ABF16) {
            const unsigned short* Ab = (const unsigned short*)Araw;
            #pragma unroll
            for (int i = 0; i < 2; ++i) {
                const int r = crow + 64 * i;
                const unsigned short* gsrc = &Ab[(size_t)(bM + r) * lda + k0 + chunk * 8];
                unsigned short* ldst = &As[buf][r * ST + chunk * 8];
#if defined(HAVE_ASYNC_LDS)
                __builtin_amdgcn_global_load_async_to_lds_b128(
                    (__attribute__((address_space(1))) v4i*)gsrc,
                    (__attribute__((address_space(3))) v4i*)ldst, 0, 0);
#else
                *(uint4*)ldst = *(const uint4*)gsrc;
#endif
            }
        } else {
            const float* Af = (const float*)Araw;
            #pragma unroll
            for (int i = 0; i < 4; ++i) {
                const int r = srow + 32 * i;
                const float4 v = *(const float4*)&Af[(size_t)(bM + r) * lda + k0 + slot * 4];
                uint2 p;
                p.x = pk2bf(v.x, v.y);
                p.y = pk2bf(v.z, v.w);
                *(uint2*)&As[buf][r * ST + slot * 4] = p;
            }
        }
        #pragma unroll
        for (int i = 0; i < 2; ++i) {
            const int r = crow + 64 * i;
            const unsigned short* gsrc = &Wb[(size_t)(bN + r) * K + k0 + chunk * 8];
            unsigned short* ldst = &Ws[buf][r * ST + chunk * 8];
#if defined(HAVE_ASYNC_LDS)
            __builtin_amdgcn_global_load_async_to_lds_b128(
                (__attribute__((address_space(1))) v4i*)gsrc,
                (__attribute__((address_space(3))) v4i*)ldst, 0, 0);
#else
            *(uint4*)ldst = *(const uint4*)gsrc;
#endif
        }
    };

    stage(0, 0);
    int cur = 0;

    for (int k0 = 0; k0 < K; k0 += BK) {
        const bool hasNext = (k0 + BK) < K;
        if (hasNext) stage(k0 + BK, cur ^ 1);     // pipeline next tile
        if (k0 + 2 * BK < K) {
            if constexpr (ABF16)
                __builtin_prefetch(&((const unsigned short*)Araw)[(size_t)(bM + crow) * lda + k0 + 2 * BK], 0, 1);
            else
                __builtin_prefetch(&((const float*)Araw)[(size_t)(bM + srow) * lda + k0 + 2 * BK], 0, 1);
        }
        // allow next tile's copies in flight; require current tile complete
        if (hasNext) wait_async<NASYNC>(); else wait_async<0>();
        __syncthreads();

        // ---- fragments per CDNA5 VGPR layouts ----
        v16bf afrag[2], bfrag[4];
        #pragma unroll
        for (int mi = 0; mi < 2; ++mi) {
            const int row = wm * 32 + mi * 16 + lrow;
            const unsigned short* p = &As[cur][row * ST];
            // elems 0..7 : K = 8*half .. +7 ; elems 8..15 : K = 16+8*half .. +7
            ((v8bf*)&afrag[mi])[0] = *(const v8bf*)&p[8 * half];
            ((v8bf*)&afrag[mi])[1] = *(const v8bf*)&p[16 + 8 * half];
        }
        #pragma unroll
        for (int ni = 0; ni < 4; ++ni) {
            const int col = wn * 64 + ni * 16 + lrow;
            const unsigned short* p = &Ws[cur][col * ST];
            // B: elems 0..7 : K = 8*half .. +7 ; elems 8..15 : K = 16+8*half .. +7
            ((v8bf*)&bfrag[ni])[0] = *(const v8bf*)&p[8 * half];
            ((v8bf*)&bfrag[ni])[1] = *(const v8bf*)&p[16 + 8 * half];
        }

        #pragma unroll
        for (int mi = 0; mi < 2; ++mi)
            #pragma unroll
            for (int ni = 0; ni < 4; ++ni)
                acc[mi][ni] = __builtin_amdgcn_wmma_f32_16x16x32_bf16(
                    false, afrag[mi], false, bfrag[ni],
                    (short)0, acc[mi][ni], false, false);
        __syncthreads();
        cur ^= 1;
    }

    // ---- epilogue + store (C/D layout: col = lane&15, row = r + 8*half) ----
    #pragma unroll
    for (int mi = 0; mi < 2; ++mi)
        #pragma unroll
        for (int ni = 0; ni < 4; ++ni) {
            const int col = bN + wn * 64 + ni * 16 + lrow;
            #pragma unroll
            for (int r = 0; r < 8; ++r) {
                const int row = bM + wm * 32 + mi * 16 + r + 8 * half;
                float v = acc[mi][ni][r];
                if constexpr (EPI == EPI_BIAS_SOFTPLUS) {
                    v += aux[col];
                    v = (v > 20.f) ? v : log1pf(__expf(v));
                } else if constexpr (EPI == EPI_RELU) {
                    v = fmaxf(v, 0.f);
                } else if constexpr (EPI == EPI_ADD_RES) {
                    v += aux[(size_t)row * N + col];
                }
                if constexpr (OUTBF16)
                    ((unsigned short*)outRaw)[(size_t)row * N + col] = bf16bits(v);
                else
                    ((float*)outRaw)[(size_t)row * N + col] = v;
            }
        }
}

// ---------------- LayerNorm: one block per row, bf16 output ----------------
__global__ __launch_bounds__(256)
void layernorm_k(const float* __restrict__ x, const float* __restrict__ g,
                 const float* __restrict__ b, unsigned short* __restrict__ out, int D)
{
    __shared__ float sbuf[256];
    const int row = blockIdx.x;
    const float* xr = x + (size_t)row * D;

    float lsum = 0.f;
    for (int i = threadIdx.x; i < D; i += 256) lsum += xr[i];
    sbuf[threadIdx.x] = lsum; __syncthreads();
    for (int s = 128; s > 0; s >>= 1) {
        if (threadIdx.x < s) sbuf[threadIdx.x] += sbuf[threadIdx.x + s];
        __syncthreads();
    }
    const float mu = sbuf[0] / D;
    __syncthreads();

    float lvar = 0.f;
    for (int i = threadIdx.x; i < D; i += 256) { float d = xr[i] - mu; lvar += d * d; }
    sbuf[threadIdx.x] = lvar; __syncthreads();
    for (int s = 128; s > 0; s >>= 1) {
        if (threadIdx.x < s) sbuf[threadIdx.x] += sbuf[threadIdx.x + s];
        __syncthreads();
    }
    const float rstd = rsqrtf(sbuf[0] / D + 1e-5f);

    for (int i = threadIdx.x; i < D; i += 256)
        out[(size_t)row * D + i] = bf16bits((xr[i] - mu) * rstd * g[i] + b[i]);
}

// ---------------- depthwise causal conv (D_CONV=4) + SiLU ----------------
// writes f32 (for scan) and bf16 (for x-proj GEMM)
__global__ __launch_bounds__(256)
void conv_silu_k(const float* __restrict__ xz,
                 const float* __restrict__ cw, const float* __restrict__ cb,
                 float* __restrict__ xc, unsigned short* __restrict__ xcbf)
{
    const int idx = blockIdx.x * 256 + threadIdx.x;
    if (idx >= kB * kL * kDI) return;
    const int d = idx % kDI;
    const int l = (idx / kDI) % kL;
    const int b = idx / (kDI * kL);

    float acc = cb[d];
    #pragma unroll
    for (int k = 0; k < 4; ++k) {
        const int ls = l - 3 + k;
        if (ls >= 0)
            acc += cw[d * 4 + k] * xz[((size_t)(b * kL + ls)) * (2 * kDI) + d];
    }
    const float s = acc / (1.f + __expf(-acc));   // SiLU
    xc[idx] = s;
    xcbf[idx] = bf16bits(s);
}

// ---------------- selective scan (fused D-skip + SiLU(z) gate) ------------
__global__ __launch_bounds__(256)
void scan_k(const float* __restrict__ dt, const float* __restrict__ dbl,
            const float* __restrict__ xc,      // u (f32)
            const float* __restrict__ xz,      // z = xz[..., D_INNER:]
            const float* __restrict__ A_log, const float* __restrict__ Dp,
            unsigned short* __restrict__ ybf)  // gated y (bf16, GEMM input)
{
    __shared__ float sBC[32];                  // [0:16)=B_t, [16:32)=C_t
    const int b = blockIdx.x / (kDI / 256);
    const int d = (blockIdx.x % (kDI / 256)) * 256 + threadIdx.x;

    float A[kDS], h[kDS];
    #pragma unroll
    for (int n = 0; n < kDS; ++n) {
        A[n] = -__expf(A_log[d * kDS + n]);
        h[n] = 0.f;
    }
    const float Dd = Dp[d];

    for (int l = 0; l < kL; ++l) {
        const size_t base = (size_t)b * kL + l;
        if (threadIdx.x < 32)
            sBC[threadIdx.x] = dbl[base * kNP + kDTR + threadIdx.x];
        __syncthreads();

        const float dtv = dt[base * kDI + d];
        const float u   = xc[base * kDI + d];
        const float du  = dtv * u;
        float y = 0.f;
        #pragma unroll
        for (int n = 0; n < kDS; ++n) {
            h[n] = __expf(dtv * A[n]) * h[n] + du * sBC[n];
            y += h[n] * sBC[16 + n];
        }
        const float z  = xz[base * (2 * kDI) + kDI + d];
        const float sz = z / (1.f + __expf(-z));
        ybf[base * kDI + d] = bf16bits((y + u * Dd) * sz);
        __syncthreads();
    }
}

// ---------------- workspace layout (bytes) ----------------
static constexpr size_t SZ_WBIN = (size_t)kDF * kDM * 2;            // W_in bf16 (4096x1024)
static constexpr size_t SZ_WBXP = (size_t)kNP * kDI * 2;            // x-proj padded to 128 rows
static constexpr size_t SZ_WBDT = (size_t)kDI * kDTR * 2;
static constexpr size_t SZ_WBO  = (size_t)kDM * kDI * 2;
static constexpr size_t SZ_WBF1 = (size_t)kDF * kDM * 2;
static constexpr size_t SZ_WBF2 = (size_t)kDM * kDF * 2;
static constexpr size_t SZ_UBF  = (size_t)kBL * kDM * 2;            // u / hn bf16
static constexpr size_t SZ_XZ   = (size_t)kBL * 2 * kDI * 4;        // xz f32 (later: relu bf16)
static constexpr size_t SZ_XC   = (size_t)kBL * kDI * 4;            // xc f32
static constexpr size_t SZ_XCBF = (size_t)kBL * kDI * 2;            // xc bf16 (later: y bf16)
static constexpr size_t SZ_DT   = (size_t)kBL * kDI * 4;
static constexpr size_t SZ_DBL  = (size_t)kBL * kNP * 4;            // dbl padded to 128 cols
static constexpr size_t SZ_H    = (size_t)kBL * kDM * 4;

static constexpr size_t O_WBIN = 0;
static constexpr size_t O_WBXP = O_WBIN + SZ_WBIN;
static constexpr size_t O_WBDT = O_WBXP + SZ_WBXP;
static constexpr size_t O_WBO  = O_WBDT + SZ_WBDT;
static constexpr size_t O_WBF1 = O_WBO  + SZ_WBO;
static constexpr size_t O_WBF2 = O_WBF1 + SZ_WBF1;
static constexpr size_t O_UBF  = O_WBF2 + SZ_WBF2;
static constexpr size_t O_XZ   = O_UBF  + SZ_UBF;
static constexpr size_t O_XC   = O_XZ   + SZ_XZ;
static constexpr size_t O_XCBF = O_XC   + SZ_XC;
static constexpr size_t O_DT   = O_XCBF + SZ_XCBF;
static constexpr size_t O_DBL  = O_DT   + SZ_DT;
static constexpr size_t O_H    = O_DBL  + SZ_DBL;

extern "C" void kernel_launch(void* const* d_in, const int* in_sizes, int n_in,
                              void* d_out, int out_size, void* d_ws, size_t ws_size,
                              hipStream_t stream)
{
    const float* x        = (const float*)d_in[0];
    const float* ln1_g    = (const float*)d_in[1];
    const float* ln1_b    = (const float*)d_in[2];
    const float* ln2_g    = (const float*)d_in[3];
    const float* ln2_b    = (const float*)d_in[4];
    const float* W_in     = (const float*)d_in[5];
    const float* conv_w   = (const float*)d_in[6];
    const float* conv_b   = (const float*)d_in[7];
    const float* W_xproj  = (const float*)d_in[8];
    const float* W_dtproj = (const float*)d_in[9];
    const float* b_dtproj = (const float*)d_in[10];
    const float* A_log    = (const float*)d_in[11];
    const float* Dvec     = (const float*)d_in[12];
    const float* W_out    = (const float*)d_in[13];
    const float* ffn_w1   = (const float*)d_in[14];
    const float* ffn_w2   = (const float*)d_in[15];

    char* ws = (char*)d_ws;
    unsigned short* wb_in = (unsigned short*)(ws + O_WBIN);
    unsigned short* wb_xp = (unsigned short*)(ws + O_WBXP);
    unsigned short* wb_dt = (unsigned short*)(ws + O_WBDT);
    unsigned short* wb_o  = (unsigned short*)(ws + O_WBO);
    unsigned short* wb_f1 = (unsigned short*)(ws + O_WBF1);
    unsigned short* wb_f2 = (unsigned short*)(ws + O_WBF2);
    unsigned short* ubf   = (unsigned short*)(ws + O_UBF);   // u, later hn
    float*          xz    = (float*)(ws + O_XZ);             // later relu bf16
    unsigned short* rbf   = (unsigned short*)(ws + O_XZ);
    float*          xc    = (float*)(ws + O_XC);
    unsigned short* xcbf  = (unsigned short*)(ws + O_XCBF);  // later ybf
    unsigned short* ybf   = (unsigned short*)(ws + O_XCBF);
    float*          dt    = (float*)(ws + O_DT);
    float*          dbl   = (float*)(ws + O_DBL);            // (BL, 128) padded
    float*          h     = (float*)(ws + O_H);
    float*          out   = (float*)d_out;

    const dim3 blk(256);
    auto cvtN = [](int n) { return (n / 4 + 255) / 256; };

    // 0) weights -> bf16 (x-proj zero-padded to 128 rows)
    hipMemsetAsync(wb_xp, 0, SZ_WBXP, stream);
    cvt_bf16_k<<<cvtN(kDF * kDM), blk, 0, stream>>>(W_in,     wb_in, kDF * kDM);
    cvt_bf16_k<<<cvtN(96 * kDI),  blk, 0, stream>>>(W_xproj,  wb_xp, 96 * kDI);
    cvt_bf16_k<<<cvtN(kDI * kDTR),blk, 0, stream>>>(W_dtproj, wb_dt, kDI * kDTR);
    cvt_bf16_k<<<cvtN(kDM * kDI), blk, 0, stream>>>(W_out,    wb_o,  kDM * kDI);
    cvt_bf16_k<<<cvtN(kDF * kDM), blk, 0, stream>>>(ffn_w1,   wb_f1, kDF * kDM);
    cvt_bf16_k<<<cvtN(kDM * kDF), blk, 0, stream>>>(ffn_w2,   wb_f2, kDM * kDF);

    // 1) LN1 -> u (bf16)
    layernorm_k<<<kBL, blk, 0, stream>>>(x, ln1_g, ln1_b, ubf, kDM);
    // 2) xz = u @ W_in^T            (4096 x 4096 x 1024)
    gemm_wmma<EPI_NONE, true, false><<<dim3((2 * kDI) / BN, kBL / BM), blk, 0, stream>>>(
        ubf, kDM, wb_in, nullptr, xz, kBL, 2 * kDI, kDM);
    // 3) xc = silu(conv(xz[:, :, :D_INNER]))  (f32 + bf16)
    conv_silu_k<<<(kBL * kDI) / 256, blk, 0, stream>>>(xz, conv_w, conv_b, xc, xcbf);
    // 4) dbl = xc @ W_xproj^T       (4096 x 128pad x 2048)
    gemm_wmma<EPI_NONE, true, false><<<dim3(kNP / BN, kBL / BM), blk, 0, stream>>>(
        xcbf, kDI, wb_xp, nullptr, dbl, kBL, kNP, kDI);
    // 5) dt = softplus(dbl[:, :, :64] @ W_dtproj^T + b)   (4096 x 2048 x 64)
    gemm_wmma<EPI_BIAS_SOFTPLUS, false, false><<<dim3(kDI / BN, kBL / BM), blk, 0, stream>>>(
        dbl, kNP, wb_dt, b_dtproj, dt, kBL, kDI, kDTR);
    // 6) selective scan + D-skip + silu(z) gate -> ybf (overwrites xcbf)
    scan_k<<<kB * (kDI / 256), blk, 0, stream>>>(dt, dbl, xc, xz, A_log, Dvec, ybf);
    // 7) h = y @ W_out^T + x        (4096 x 1024 x 2048)
    gemm_wmma<EPI_ADD_RES, true, false><<<dim3(kDM / BN, kBL / BM), blk, 0, stream>>>(
        ybf, kDI, wb_o, x, h, kBL, kDM, kDI);
    // 8) LN2 -> hn (reuse ubf)
    layernorm_k<<<kBL, blk, 0, stream>>>(h, ln2_g, ln2_b, ubf, kDM);
    // 9) relu(hn @ ffn_w1^T) -> bf16 (4096 x 4096 x 1024), reuse xz region
    gemm_wmma<EPI_RELU, true, true><<<dim3(kDF / BN, kBL / BM), blk, 0, stream>>>(
        ubf, kDM, wb_f1, nullptr, rbf, kBL, kDF, kDM);
    // 10) out = relu_out @ ffn_w2^T + h   (4096 x 1024 x 4096)
    gemm_wmma<EPI_ADD_RES, true, false><<<dim3(kDM / BN, kBL / BM), blk, 0, stream>>>(
        rbf, kDF, wb_f2, h, out, kBL, kDM, kDF);
}
